// VQModule_19207093748585
// MI455X (gfx1250) — matrix operand order repaired and checked by reading, not compiled, not run on hardware
//
#include <hip/hip_runtime.h>
#include <hip/hip_bf16.h>

typedef __attribute__((ext_vector_type(16))) _Float16 v16h;
typedef __attribute__((ext_vector_type(8)))  float    v8f;

#define BATCH   8
#define NTOK    4096
#define DDIM    256
#define KCB     8192
#define NT      (KCB / 16)            /* 512 code tiles */
#define M_TOTAL (BATCH * NTOK)        /* 32768 tokens  */
#define BND     (M_TOTAL * DDIM)      /* 8388608 qe elements */
#define CB_SCALE 8192.0f
#define LDS_ROW 272                    /* 256 halfs + 16 pad -> 544B rows (32B aligned) */
#define LDS_TILE (16 * LDS_ROW)        /* halfs per buffer (8704 B) */

// ---------------------------------------------------------------------------
// Kernel 1: convert codebook f32 -> f16 (scaled), compute scaled norms,
//           zero the commit-loss output slot.
// grid = KCB blocks x 256 threads (one block per codebook row)
// ---------------------------------------------------------------------------
__global__ void vq_prep_kernel(const float* __restrict__ cb,
                               _Float16* __restrict__ cb_h,
                               float* __restrict__ norm_s,
                               float* __restrict__ commit_slot) {
    const int row = blockIdx.x;
    const int tid = threadIdx.x;
    const float v = cb[(size_t)row * DDIM + tid];
    cb_h[(size_t)row * DDIM + tid] = (_Float16)(v * CB_SCALE);

    float s = v * v;
    #pragma unroll
    for (int off = 16; off >= 1; off >>= 1) s += __shfl_xor(s, off, 32);
    __shared__ float warr[8];
    const int lane = tid & 31, w = tid >> 5;
    if (lane == 0) warr[w] = s;
    __syncthreads();
    if (tid == 0) {
        float t = 0.0f;
        #pragma unroll
        for (int i = 0; i < 8; ++i) t += warr[i];
        norm_s[row] = CB_SCALE * t;   // scaled ||cb_k||^2, matches scaled dot
    }
    if (row == 0 && tid == 0) *commit_slot = 0.0f;
}

// ---------------------------------------------------------------------------
// helper: load 8 contiguous f32 and convert into v16h elements [base, base+8)
// ---------------------------------------------------------------------------
__device__ __forceinline__ void cvt8(const float* __restrict__ p, v16h& dst, int base) {
    float4 f0 = ((const float4*)p)[0];
    float4 f1 = ((const float4*)p)[1];
    dst[base + 0] = (_Float16)f0.x; dst[base + 1] = (_Float16)f0.y;
    dst[base + 2] = (_Float16)f0.z; dst[base + 3] = (_Float16)f0.w;
    dst[base + 4] = (_Float16)f1.x; dst[base + 5] = (_Float16)f1.y;
    dst[base + 6] = (_Float16)f1.z; dst[base + 7] = (_Float16)f1.w;
}

// ---------------------------------------------------------------------------
// Kernel 2: fused GEMM + argmin.
// grid = M_TOTAL/256 = 128 blocks x 256 threads (8 waves).
// Wave w owns 2 token tiles (32 tokens). Block sweeps all 8192 codes in
// 16-code tiles staged through double-buffered LDS (shared by the 8 waves).
// ---------------------------------------------------------------------------
__launch_bounds__(256, 1)
__global__ void vq_argmin_kernel(const float* __restrict__ x,
                                 const _Float16* __restrict__ cb_h,
                                 const float* __restrict__ norm_s,
                                 int* __restrict__ idx_ws,
                                 float* __restrict__ out_idx_f) {
    __shared__ alignas(32) _Float16 lds[2 * LDS_TILE];   // 17408 bytes

    const int tid  = threadIdx.x;
    const int w    = tid >> 5;
    const int lane = tid & 31;
    const int h    = lane >> 4;     // half-wave id
    const int m    = lane & 15;     // lane-in-half
    const int blockTokBase = blockIdx.x * 256;

    // ---- load A fragments (2 token tiles x 8 K32-chunks), contiguous K = 16h + 0..15 per half ----
    v16h a[2][8];
    #pragma unroll
    for (int t = 0; t < 2; ++t) {
        const int row = blockTokBase + w * 32 + t * 16 + m;
        const float* xr = x + (size_t)row * DDIM;
        #pragma unroll
        for (int c = 0; c < 8; ++c) {
            cvt8(xr + 32 * c + 16 * h,      a[t][c], 0);   // K = 16h + 0..7
            cvt8(xr + 32 * c + 16 * h + 8,  a[t][c], 8);   // K = 16h + 8..15
        }
    }

    // running argmax of (dot_scaled - 0.5*norm_scaled)  == argmin of score.
    // Strict '>' update + monotonically increasing code index == first-index
    // tie behavior, with no branches (keeps EXEC all-ones around WMMA).
    float maxv[2][8];
    int   maxi[2][8];
    #pragma unroll
    for (int t = 0; t < 2; ++t)
        #pragma unroll
        for (int r = 0; r < 8; ++r) { maxv[t][r] = -3.4e38f; maxi[t][r] = 0; }

    // ---- codebook tile streaming: global -> regs -> LDS (double buffered) ----
    const uint4* gsrc = (const uint4*)cb_h;    // row stride = 32 uint4 (512B)
    uint4* ldsv = (uint4*)lds;
    const int r0 = tid >> 5,         c0 = tid & 31;   // j = tid        (rows 0..7)
    const int r1 = (tid + 256) >> 5, c1 = tid & 31;   // j = tid + 256  (rows 8..15)
    const int d0 = r0 * (LDS_ROW / 8) + c0;           // uint4 offset in a buffer
    const int d1 = r1 * (LDS_ROW / 8) + c1;

    // stage tile 0 into buffer 0
    uint4 pf0 = gsrc[(size_t)r0 * 32 + c0];
    uint4 pf1 = gsrc[(size_t)r1 * 32 + c1];
    ldsv[d0] = pf0;
    ldsv[d1] = pf1;
    // prefetch tile 1 into registers
    pf0 = gsrc[((size_t)16 + r0) * 32 + c0];
    pf1 = gsrc[((size_t)16 + r1) * 32 + c1];
    float nv = norm_s[m];                       // norm for tile 0
    __syncthreads();

    for (int nt = 0; nt < NT; ++nt) {
        // commit prefetched tile nt+1 into the other buffer
        if (nt + 1 < NT) {
            uint4* dst = ldsv + ((nt + 1) & 1) * (LDS_TILE / 8);
            dst[d0] = pf0;
            dst[d1] = pf1;
        }
        // prefetch tile nt+2 from global
        if (nt + 2 < NT) {
            pf0 = gsrc[((size_t)(nt + 2) * 16 + r0) * 32 + c0];
            pf1 = gsrc[((size_t)(nt + 2) * 16 + r1) * 32 + c1];
        }
        // prefetch next tile's norm
        const float nv_next = (nt + 1 < NT) ? norm_s[(nt + 1) * 16 + m] : 0.0f;

        const int   code = nt * 16 + m;            // this lane's code column
        const float bias = -0.5f * nv;             // fold norm into C init
        v8f acc0 = {bias, bias, bias, bias, bias, bias, bias, bias};
        v8f acc1 = acc0;

        const _Float16* buf = lds + (nt & 1) * LDS_TILE;
        #pragma unroll
        for (int c = 0; c < 8; ++c) {
            // B fragment, dense B layout: lane n=m, K = 16h + 0..15 contiguous
            v16h b = *(const v16h*)&buf[m * LDS_ROW + 32 * c + 16 * h];
            acc0 = __builtin_amdgcn_wmma_f32_16x16x32_f16(
                       false, a[0][c], false, b, (short)0, acc0, false, false);
            acc1 = __builtin_amdgcn_wmma_f32_16x16x32_f16(
                       false, a[1][c], false, b, (short)0, acc1, false, false);
        }

        #pragma unroll
        for (int r = 0; r < 8; ++r) {
            const float s0 = acc0[r];
            const bool  b0 = s0 > maxv[0][r];
            maxv[0][r] = b0 ? s0   : maxv[0][r];
            maxi[0][r] = b0 ? code : maxi[0][r];
            const float s1 = acc1[r];
            const bool  b1 = s1 > maxv[1][r];
            maxv[1][r] = b1 ? s1   : maxv[1][r];
            maxi[1][r] = b1 ? code : maxi[1][r];
        }

        nv = nv_next;
        __syncthreads();
    }

    // ---- cross-lane argmax within each 16-lane half (first index on ties) ----
    #pragma unroll
    for (int t = 0; t < 2; ++t) {
        #pragma unroll
        for (int r = 0; r < 8; ++r) {
            float v = maxv[t][r];
            int   i = maxi[t][r];
            #pragma unroll
            for (int off = 8; off >= 1; off >>= 1) {
                const float ov = __shfl_xor(v, off, 16);
                const int   oi = __shfl_xor(i, off, 16);
                const bool take = (ov > v) || (ov == v && oi < i);
                v = take ? ov : v;
                i = take ? oi : i;
            }
            maxv[t][r] = v; maxi[t][r] = i;
        }
    }

    // C-matrix row mapping: VGPR r, half h -> token M = 8h + r
    if (m == 0) {
        #pragma unroll
        for (int t = 0; t < 2; ++t) {
            #pragma unroll
            for (int r = 0; r < 8; ++r) {
                const int token = blockTokBase + w * 32 + t * 16 + 8 * h + r;
                idx_ws[token]    = maxi[t][r];
                out_idx_f[token] = (float)maxi[t][r];
            }
        }
    }
}

// ---------------------------------------------------------------------------
// Kernel 3: gather qe = codebook[idx] (straight-through forward value) and
//           accumulate commit_loss = mean((x - q)^2).
// grid = M_TOTAL blocks x 256 threads (one block per token)
// ---------------------------------------------------------------------------
__global__ void vq_gather_kernel(const float* __restrict__ x,
                                 const float* __restrict__ cb,
                                 const int* __restrict__ idx_ws,
                                 float* __restrict__ qe,
                                 float* __restrict__ commit_slot) {
    const int token = blockIdx.x;
    const int tid   = threadIdx.x;
    const int idx   = idx_ws[token];

    const float q  = cb[(size_t)idx * DDIM + tid];
    const float xv = x[(size_t)token * DDIM + tid];
    qe[(size_t)token * DDIM + tid] = q;

    const float d = xv - q;
    float s = d * d;
    #pragma unroll
    for (int off = 16; off >= 1; off >>= 1) s += __shfl_xor(s, off, 32);
    __shared__ float warr[8];
    const int lane = tid & 31, w = tid >> 5;
    if (lane == 0) warr[w] = s;
    __syncthreads();
    if (tid == 0) {
        float t = 0.0f;
        #pragma unroll
        for (int i = 0; i < 8; ++i) t += warr[i];
        atomicAdd(commit_slot, t * (1.0f / (float)BND));
    }
}

// ---------------------------------------------------------------------------
extern "C" void kernel_launch(void* const* d_in, const int* in_sizes, int n_in,
                              void* d_out, int out_size, void* d_ws, size_t ws_size,
                              hipStream_t stream) {
    (void)in_sizes; (void)n_in; (void)out_size; (void)ws_size;

    const float* x  = (const float*)d_in[0];   // [8, 4096, 256] f32
    const float* cb = (const float*)d_in[1];   // [8192, 256] f32

    float* out       = (float*)d_out;
    float* qe        = out;                    // BND floats
    float* commit    = out + BND;              // 1 float
    float* out_idx_f = out + BND + 1;          // M_TOTAL floats

    // workspace layout
    _Float16* cb_h   = (_Float16*)d_ws;                                  // 4 MB
    float*    norm_s = (float*)((char*)d_ws + (size_t)KCB * DDIM * 2);   // 32 KB
    int*      idx_ws = (int*)((char*)d_ws + (size_t)KCB * DDIM * 2
                                          + (size_t)KCB * 4);            // 128 KB

    vq_prep_kernel  <<<KCB,           256, 0, stream>>>(cb, cb_h, norm_s, commit);
    vq_argmin_kernel<<<M_TOTAL / 256, 256, 0, stream>>>(x, cb_h, norm_s, idx_ws, out_idx_f);
    vq_gather_kernel<<<M_TOTAL,       256, 0, stream>>>(x, cb, idx_ws, qe, commit);
}
